// TemporalSAGE_20615843021268
// MI455X (gfx1250) — matrix operand order repaired and checked by reading, not compiled, not run on hardware
//
#include <hip/hip_runtime.h>
#include <hip/hip_bf16.h>

// ---------------------------------------------------------------------------
// TemporalSAGE forward for MI455X (gfx1250, wave32, WMMA).
//   B=2, W=32, N=768, F=4, U=32, G=4U=128, KH=2 hops, K=8 rel channels, INP=72
// ---------------------------------------------------------------------------

typedef __attribute__((ext_vector_type(16))) _Float16 v16h;
typedef __attribute__((ext_vector_type(8)))  float    v8f;

#define WMMA_F16(a, b, c) \
  __builtin_amdgcn_wmma_f32_16x16x32_f16(false, (a), false, (b), (short)0, (c), false, false)

// TRANS-pipe math: v_exp_f32 + v_rcp_f32 co-execute with WMMA (XDL) ops.
__device__ __forceinline__ float fast_rcp(float x) { return __builtin_amdgcn_rcpf(x); }
__device__ __forceinline__ float sigm(float x) {
  return fast_rcp(1.f + __expf(-x));            // x->-inf: rcp(inf)=0; x->+inf: 1
}
__device__ __forceinline__ float tanh_fast(float x) {
  return 1.f - 2.f * fast_rcp(__expf(2.f * x) + 1.f);  // safe at +/-inf
}
__device__ __forceinline__ float lrelu(float x) { return x > 0.f ? x : 0.01f * x; }

// ---------------------------------------------------------------------------
// Kernel 1: per-stock LSTM; writes last hidden state to seq & e.
// One wave = 16 sequences; 8 waves/block; 12 blocks cover M = 1536.
// Per step: z(16x128) = bias + x_t(16x4)@Wih^T (VALU, folded into C) +
//           h(16x32)@Whh^T via 8x v_wmma_f32_16x16x32_f16 (K=32 == U).
// h is transposed C-layout -> A-layout through a private per-wave LDS slab
// (intra-wave LDS ops are in-order; no barriers needed).
// ---------------------------------------------------------------------------
__global__ __launch_bounds__(256) void lstm_kernel(
    const float* __restrict__ x,     // (B,W,N,F)
    const float* __restrict__ Wih,   // (128,4)
    const float* __restrict__ Whh,   // (128,32)
    const float* __restrict__ bih,   // (128)
    const float* __restrict__ bhh,   // (128)
    float* __restrict__ seq,         // (B*N,32)
    float* __restrict__ eout)        // (B*N,32)
{
  constexpr int Wt = 32, Nn = 768, U = 32;
  __shared__ float hbuf[8][16 * 32];           // 16KB: per-wave h transpose slab
  const int lane = threadIdx.x & 31, wave = threadIdx.x >> 5;
  const int lc = lane & 15, half = lane >> 4;
  const int rowTile = (blockIdx.x * 8 + wave) * 16;   // base m in [0,1536)
  const int b = rowTile / Nn;
  const int nBase = rowTile % Nn;
  float* hl = hbuf[wave];

#pragma unroll
  for (int idx = 0; idx < 16; ++idx) hl[idx * 32 + lane] = 0.f;   // h0 = 0

  // Loop-invariant per-lane data. C-layout column for tile ct: col = ct*16+lc.
  v16h Bw[8]; float4 wih[8]; float bias[8];
#pragma unroll
  for (int ct = 0; ct < 8; ++ct) {
    const int col = ct * 16 + lc;
    const int kb = half * 16;            // B layout: lanes<16 K=0..15, >=16 K=16..31
    v16h bw;
#pragma unroll
    for (int kk = 0; kk < 16; ++kk) bw[kk] = (_Float16)Whh[col * U + kb + kk];
    Bw[ct]   = bw;
    wih[ct]  = *(const float4*)(Wih + col * 4);
    bias[ct] = bih[col] + bhh[col];
  }

  v8f cst[2];
#pragma unroll
  for (int r = 0; r < 8; ++r) { cst[0][r] = 0.f; cst[1][r] = 0.f; }

  for (int t = 0; t < Wt; ++t) {
    // x rows for this wave (row M = r + half*8)
    float4 xr[8];
#pragma unroll
    for (int r = 0; r < 8; ++r) {
      const int n = nBase + r + half * 8;
      xr[r] = *(const float4*)(x + ((size_t)(b * Wt + t) * Nn + n) * 4);
    }
    // A operand (f16 16x32): lanes<16: M=lc, K=0..15; lanes>=16: K=16..31
    v16h a;
    {
      const int k0 = half * 16, k1 = half * 16 + 8;
#pragma unroll
      for (int e2 = 0; e2 < 8; ++e2) {
        a[e2]     = (_Float16)hl[lc * U + k0 + e2];
        a[8 + e2] = (_Float16)hl[lc * U + k1 + e2];
      }
    }
    // z = bias + x@Wih^T (VALU) + h@Whh^T (WMMA), 8 gate-column tiles
    v8f z[8];
#pragma unroll
    for (int ct = 0; ct < 8; ++ct) {
      v8f acc;
#pragma unroll
      for (int r = 0; r < 8; ++r)
        acc[r] = bias[ct] + xr[r].x * wih[ct].x + xr[r].y * wih[ct].y +
                 xr[r].z * wih[ct].z + xr[r].w * wih[ct].w;
      z[ct] = WMMA_F16(a, Bw[ct], acc);
    }
    // gates (i,f,g,o = tiles 0-1, 2-3, 4-5, 6-7); fully unrolled (no movrel)
#pragma unroll
    for (int ht = 0; ht < 2; ++ht) {
#pragma unroll
      for (int r = 0; r < 8; ++r) {
        const float iv = sigm(z[0 + ht][r]);
        const float fv = sigm(z[2 + ht][r]);
        const float gv = tanh_fast(z[4 + ht][r]);
        const float ov = sigm(z[6 + ht][r]);
        const float cc = fv * cst[ht][r] + iv * gv;
        cst[ht][r] = cc;
        hl[(r + half * 8) * U + ht * 16 + lc] = ov * tanh_fast(cc);
      }
    }
  }
#pragma unroll
  for (int rr = 0; rr < 16; ++rr) {
    const int m = rowTile + rr;
    const float v = hl[rr * 32 + lane];
    seq[m * U + lane]  = v;
    eout[m * U + lane] = v;
  }
}

// ---------------------------------------------------------------------------
// Kernel 2: degrees D[i] = sum_j mask[i,j], mask = (sum_k enc[i,j,k] > 0)
// ---------------------------------------------------------------------------
__global__ void deg_kernel(const float* __restrict__ enc, float* __restrict__ Dg)
{
  const int i = blockIdx.x * blockDim.x + threadIdx.x;
  if (i >= 768) return;
  float cnt = 0.f;
  for (int j = 0; j < 768; ++j) {
    const float* ep = enc + ((size_t)i * 768 + j) * 8;
    const float4 e0 = *(const float4*)ep;
    const float4 e1 = *(const float4*)(ep + 4);
    const float s = e0.x + e0.y + e0.z + e0.w + e1.x + e1.y + e1.z + e1.w;
    cnt += (s > 0.f) ? 1.f : 0.f;
  }
  Dg[i] = cnt;
}

// ---------------------------------------------------------------------------
// Kernel 3: xi = e @ W1[:32], xj = e @ W1[32:64]   (B*N x 72 each)
// ---------------------------------------------------------------------------
__global__ void xixj_kernel(const float* __restrict__ e, const float* __restrict__ W1,
                            float* __restrict__ xi, float* __restrict__ xj)
{
  const int idx = blockIdx.x * blockDim.x + threadIdx.x;
  if (idx >= 2 * 768 * 72) return;
  const int p = idx % 72, row = idx / 72;
  const float* er = e + row * 32;
  float si = 0.f, sj = 0.f;
#pragma unroll
  for (int u = 0; u < 32; ++u) {
    const float ev = er[u];
    si += ev * W1[u * 72 + p];
    sj += ev * W1[(32 + u) * 72 + p];
  }
  xi[idx] = si;
  xj[idx] = sj;
}

// ---------------------------------------------------------------------------
// Kernel 4: fused relational hop. Never materializes the (B,N,N,72) tensor.
// Grid: x = b*8 + iChunk (16), y = jTile (6). Each wave owns 16 j's, loops
// 96 i's. Per i: xe tile via 5x v_wmma_f32_16x16x32_f16 (A = 16x8 enc rows
// zero-padded to K=32; B = W1 rows 64..71 zero-padded), then the masked
// leaky/dot-w2 stage in VALU + 16-lane butterfly reduction. Partial column
// sums go to part[(b*8+iChunk), j] -> deterministic reduce in eupd_kernel
// (no float atomics -> bitwise-reproducible across graph replays).
// ---------------------------------------------------------------------------
__global__ __launch_bounds__(256) void rel_kernel(
    const float* __restrict__ enc,   // (N,N,8)
    const float* __restrict__ xi,    // (B,N,72)
    const float* __restrict__ xj,    // (B,N,72)
    const float* __restrict__ Dg,    // (N)
    const float* __restrict__ W1,    // (72,72) for this hop
    const float* __restrict__ b1,    // (72)
    const float* __restrict__ w2,    // (72)
    const float* __restrict__ b2p,   // &hop_b2[k]
    float* __restrict__ part)        // (B*8, N)
{
  constexpr int Nn = 768, INP = 72;
  const int lane = threadIdx.x & 31, wave = threadIdx.x >> 5;
  const int lc = lane & 15, half = lane >> 4;
  const int bc = blockIdx.x;                 // b*8 + iChunk
  const int b = bc >> 3, iChunk = bc & 7;
  const int iBase = iChunk * 96;
  const int jBase = blockIdx.y * 128 + wave * 16;
  const float b2 = *b2p;

  // per-lane columns (5 tiles cover 72, tile 4 half-valid; invalid cols -> 0)
  int cols[5]; float w2v[5], b1v[5];
#pragma unroll
  for (int ct = 0; ct < 5; ++ct) {
    const int c = ct * 16 + lc;
    const bool ok = (c < INP);
    cols[ct] = ok ? c : 0;
    w2v[ct] = ok ? w2[c] : 0.f;
    b1v[ct] = ok ? b1[c] : 0.f;
  }
  // B operand: W1 rows 64..71 (enc channels), K padded 8 -> 32 with zeros
  v16h Bw[5];
#pragma unroll
  for (int ct = 0; ct < 5; ++ct) {
    v16h bw;
#pragma unroll
    for (int kk = 0; kk < 16; ++kk) bw[kk] = (_Float16)0.f;
    const int c = ct * 16 + lc;
    if (half == 0 && c < INP) {
#pragma unroll
      for (int kk = 0; kk < 8; ++kk) bw[kk] = (_Float16)W1[(64 + kk) * INP + c];
    }
    Bw[ct] = bw;
  }
  // hoist xj[b, j, col] (loop-invariant over i)
  float xjv[5][8];
#pragma unroll
  for (int ct = 0; ct < 5; ++ct)
#pragma unroll
    for (int r = 0; r < 8; ++r) {
      const int j = jBase + r + half * 8;
      xjv[ct][r] = (ct * 16 + lc < INP) ? xj[((size_t)b * Nn + j) * INP + cols[ct]] : 0.f;
    }

  float acc[8];
#pragma unroll
  for (int r = 0; r < 8; ++r) acc[r] = 0.f;

  for (int ii = 0; ii < 96; ++ii) {
    const int i = iBase + ii;
    // A operand: 16 j-rows of enc[i,j,0..7] in lanes 0..15 (K padded to 32)
    v16h a;
#pragma unroll
    for (int kk = 0; kk < 16; ++kk) a[kk] = (_Float16)0.f;
    bool mb = false;
    if (half == 0) {
      const float* ep = enc + ((size_t)i * Nn + (jBase + lc)) * 8;
      const float4 e0 = *(const float4*)ep;
      const float4 e1 = *(const float4*)(ep + 4);
      const float s = e0.x + e0.y + e0.z + e0.w + e1.x + e1.y + e1.z + e1.w;
      a[0] = (_Float16)e0.x; a[1] = (_Float16)e0.y;
      a[2] = (_Float16)e0.z; a[3] = (_Float16)e0.w;
      a[4] = (_Float16)e1.x; a[5] = (_Float16)e1.y;
      a[6] = (_Float16)e1.z; a[7] = (_Float16)e1.w;
      mb = (s > 0.f);
    }
    const unsigned mbits = (unsigned)__ballot(mb) & 0xFFFFu;   // mask for 16 j's
    const float dinv = fast_rcp(Dg[i]);       // D integer-valued >= 1
    float xiv[5];
#pragma unroll
    for (int ct = 0; ct < 5; ++ct)
      xiv[ct] = (ct * 16 + lc < INP) ? xi[((size_t)b * Nn + i) * INP + cols[ct]] : 0.f;

    float ps[8];
#pragma unroll
    for (int r = 0; r < 8; ++r) ps[r] = 0.f;
#pragma unroll
    for (int ct = 0; ct < 5; ++ct) {
      v8f zero;
#pragma unroll
      for (int r = 0; r < 8; ++r) zero[r] = 0.f;
      v8f xe = WMMA_F16(a, Bw[ct], zero);     // uniform flow: EXEC all ones
#pragma unroll
      for (int r = 0; r < 8; ++r) {
        const unsigned mbit = (mbits >> (r + half * 8)) & 1u;
        float pre = mbit ? (xiv[ct] + xjv[ct][r] + xe[r]) : 0.f;  // mask pre-bias term
        pre += b1v[ct];
        ps[r] += lrelu(pre) * w2v[ct];        // invalid cols contribute exactly 0
      }
    }
    // 16-lane butterfly: every lane in the group gets sum over all 72 cols
#pragma unroll
    for (int r = 0; r < 8; ++r) {
      float v = ps[r];
      v += __shfl_xor(v, 1);
      v += __shfl_xor(v, 2);
      v += __shfl_xor(v, 4);
      v += __shfl_xor(v, 8);
      const unsigned mbit = (mbits >> (r + half * 8)) & 1u;
      const float wts = lrelu(v + b2);
      acc[r] += mbit ? wts * dinv : 0.f;
    }
  }
  if (lc == 0) {
#pragma unroll
    for (int r = 0; r < 8; ++r)
      part[(size_t)bc * Nn + jBase + r + half * 8] = acc[r];
  }
}

// ---------------------------------------------------------------------------
// Kernel 5: colsum = deterministic reduce of 8 i-chunk partials; e *= colsum
// ---------------------------------------------------------------------------
__global__ void eupd_kernel(const float* __restrict__ part, float* __restrict__ e)
{
  const int idx = blockIdx.x * blockDim.x + threadIdx.x;
  if (idx >= 2 * 768 * 32) return;
  const int row = idx >> 5;            // b*768 + j
  const int b = row / 768, j = row % 768;
  float s = 0.f;
#pragma unroll
  for (int c = 0; c < 8; ++c) s += part[((size_t)(b * 8 + c)) * 768 + j];
  e[idx] *= s;
}

// ---------------------------------------------------------------------------
// Kernel 6: FC head: leaky(leaky([seq,e] @ fn_w1 + b1) @ fn_w2 + b2)
// ---------------------------------------------------------------------------
__global__ void head_kernel(const float* __restrict__ seq, const float* __restrict__ e,
                            const float* __restrict__ w1, const float* __restrict__ bv1,
                            const float* __restrict__ w2, const float* __restrict__ bv2,
                            float* __restrict__ out)
{
  const int row = blockIdx.x * blockDim.x + threadIdx.x;
  if (row >= 1536) return;
  float comb[64];
#pragma unroll
  for (int u = 0; u < 32; ++u) { comb[u] = seq[row * 32 + u]; comb[32 + u] = e[row * 32 + u]; }
  float o = bv2[0];
  for (int h = 0; h < 64; ++h) {
    float acc = bv1[h];
#pragma unroll
    for (int c = 0; c < 64; ++c) acc += comb[c] * w1[c * 64 + h];
    o += lrelu(acc) * w2[h];
  }
  out[row] = lrelu(o);
}

// ---------------------------------------------------------------------------
extern "C" void kernel_launch(void* const* d_in, const int* in_sizes, int n_in,
                              void* d_out, int out_size, void* d_ws, size_t ws_size,
                              hipStream_t stream)
{
  (void)in_sizes; (void)n_in; (void)out_size; (void)ws_size;
  const float* input_data = (const float*)d_in[0];   // (2,32,768,4)
  const float* enc        = (const float*)d_in[1];   // (768,768,8)
  const float* lstm_Wih   = (const float*)d_in[2];   // (128,4)
  const float* lstm_Whh   = (const float*)d_in[3];   // (128,32)
  const float* lstm_bih   = (const float*)d_in[4];
  const float* lstm_bhh   = (const float*)d_in[5];
  const float* hop_w1     = (const float*)d_in[6];   // (2,72,72)
  const float* hop_b1     = (const float*)d_in[7];   // (2,72)
  const float* hop_w2     = (const float*)d_in[8];   // (2,72,1)
  const float* hop_b2     = (const float*)d_in[9];   // (2,1)
  const float* fn_w1      = (const float*)d_in[10];  // (64,64)
  const float* fn_b1      = (const float*)d_in[11];
  const float* fn_w2      = (const float*)d_in[12];  // (64,1)
  const float* fn_b2      = (const float*)d_in[13];
  float* out = (float*)d_out;                        // (2,768)

  // workspace layout (floats)
  float* ws   = (float*)d_ws;
  float* seq  = ws;                 // 1536*32 = 49152
  float* e    = ws + 49152;         // 49152
  float* xi   = ws + 98304;         // 1536*72 = 110592
  float* xj   = ws + 208896;        // 110592
  float* Dg   = ws + 319488;        // 768
  float* part = ws + 320256;        // 16*768 = 12288

  lstm_kernel<<<12, 256, 0, stream>>>(input_data, lstm_Wih, lstm_Whh,
                                      lstm_bih, lstm_bhh, seq, e);
  deg_kernel<<<3, 256, 0, stream>>>(enc, Dg);

  for (int k = 0; k < 2; ++k) {
    const float* W1 = hop_w1 + (size_t)k * 72 * 72;
    xixj_kernel<<<432, 256, 0, stream>>>(e, W1, xi, xj);
    dim3 g(16, 6);
    rel_kernel<<<g, 256, 0, stream>>>(enc, xi, xj, Dg, W1,
                                      hop_b1 + k * 72, hop_w2 + k * 72,
                                      hop_b2 + k, part);
    eupd_kernel<<<192, 256, 0, stream>>>(part, e);
  }

  head_kernel<<<6, 256, 0, stream>>>(seq, e, fn_w1, fn_b1, fn_w2, fn_b2, out);
}
